// MultiHeadAttention_49813030699091
// MI455X (gfx1250) — matrix operand order repaired and checked by reading, not compiled, not run on hardware
//
#include <hip/hip_runtime.h>

#define HEAD_NUM    16
#define EMB_DIM     1024
#define PROJ_DIM    1024
#define CONTEXT_LEN 2048
#define HEAD_DIM    64
#define BATCH       2
#define M_TOTAL     (BATCH * CONTEXT_LEN)   // 4096 rows

typedef __attribute__((ext_vector_type(16))) _Float16 v16h;
typedef __attribute__((ext_vector_type(8)))  _Float16 v8h;
typedef __attribute__((ext_vector_type(8)))  float    v8f;

// ---------------------------------------------------------------------------
// Fragment helpers (CDNA5 WMMA 16x16x32 f16, wave32)
// ---------------------------------------------------------------------------
__device__ __forceinline__ v16h make_frag(v8h lo, v8h hi) {
  return __builtin_shufflevector(lo, hi, 0,1,2,3,4,5,6,7,8,9,10,11,12,13,14,15);
}

// A fragment 16x32 (MxK), row-major source. Caller passes &A[row*ld + kb]
// with row = r0 + (lane&15).
// halves 0..7  hold K = kb + sel + 0..7   (sel = 0 lanes 0-15, 8 lanes 16-31)
// halves 8..15 hold K = kb + 16 + sel + 0..7           (per ISA 7.12.2 table)
__device__ __forceinline__ v16h load_a_frag(const _Float16* row_kb, int lane) {
  const int sel = (lane & 16) ? 8 : 0;
  v8h lo = *(const v8h*)(row_kb + sel);
  v8h hi = *(const v8h*)(row_kb + 16 + sel);
  return make_frag(lo, hi);
}

// B fragment 32x16 (KxN) with B^T stored row-major [N][K]. Caller passes
// &Bt[n*ld + kb] with n = n0 + (lane&15).
// halves 0..7  hold K = kb + koff + 0..7   (koff = 0 lanes 0-15, 8 lanes 16-31)
// halves 8..15 hold K = kb + 16 + koff + 0..7   (same order as A fragment)
__device__ __forceinline__ v16h load_b_frag(const _Float16* col_kb, int lane) {
  const int koff = (lane & 16) ? 8 : 0;
  v8h lo = *(const v8h*)(col_kb + koff);
  v8h hi = *(const v8h*)(col_kb + 16 + koff);
  return make_frag(lo, hi);
}

__device__ __forceinline__ v8f wmma_f16(v16h a, v16h b, v8f c) {
  return __builtin_amdgcn_wmma_f32_16x16x32_f16(false, a, false, b,
                                                (short)0, c, false, false);
}

// ---------------------------------------------------------------------------
// Stage 0: precision conversion / weight transpose
// ---------------------------------------------------------------------------
__global__ void cvt_f32_to_f16(const float* __restrict__ in,
                               _Float16* __restrict__ out, int n) {
  int i = blockIdx.x * blockDim.x + threadIdx.x;
  if (i < n) out[i] = (_Float16)in[i];
}

// in: f32 [K=1024][N=1024] -> out: f16 [N][K]
__global__ void transpose_cvt(const float* __restrict__ in,
                              _Float16* __restrict__ out) {
  int i = blockIdx.x * blockDim.x + threadIdx.x;
  int k = i >> 10;
  int n = i & 1023;
  out[(size_t)n * EMB_DIM + k] = (_Float16)in[(size_t)k * PROJ_DIM + n];
}

// ---------------------------------------------------------------------------
// Stage 1/4: WMMA GEMM  C[m][n] = sum_k A[m][k] * Bt[n][k]
// M=4096, N=1024, K=1024. Block = 8 waves, 256x128 tile; wave = 64x64 tile.
// mode 0: f16 out, head layout [B][H][S][D]        (Q, K)
// mode 1: f16 out, transposed  [B][H][D][S]        (V)
// mode 2: f32 out, row-major [M][N]                (final output)
// ---------------------------------------------------------------------------
__global__ void __launch_bounds__(256)
gemm_wmma(const _Float16* __restrict__ A, const _Float16* __restrict__ Bt,
          _Float16* __restrict__ outh, float* __restrict__ outf, int mode) {
  const int K = EMB_DIM;
  const int lane = threadIdx.x & 31;
  const int wid  = threadIdx.x >> 5;
  const int m_base = blockIdx.x * 256 + (wid >> 1) * 64;
  const int n_base = blockIdx.y * 128 + (wid & 1) * 64;
  const int arow = lane & 15;

  v8f zero = {};
  v8f acc[4][4];
#pragma unroll
  for (int i = 0; i < 4; ++i)
#pragma unroll
    for (int j = 0; j < 4; ++j) acc[i][j] = zero;

  for (int kb = 0; kb < K; kb += 32) {
    v16h af[4];
#pragma unroll
    for (int im = 0; im < 4; ++im)
      af[im] = load_a_frag(A + (size_t)(m_base + im * 16 + arow) * K + kb, lane);
#pragma unroll
    for (int jn = 0; jn < 4; ++jn) {
      v16h bf = load_b_frag(Bt + (size_t)(n_base + jn * 16 + arow) * K + kb, lane);
#pragma unroll
      for (int im = 0; im < 4; ++im)
        acc[im][jn] = wmma_f16(af[im], bf, acc[im][jn]);
    }
  }

  // C/D layout: VGPR g, lanes 0-15 -> row g, lanes 16-31 -> row g+8; col = lane&15
  const int mrow_off = (lane & 16) ? 8 : 0;
  const int ncol = lane & 15;
#pragma unroll
  for (int im = 0; im < 4; ++im)
#pragma unroll
    for (int jn = 0; jn < 4; ++jn)
#pragma unroll
      for (int g = 0; g < 8; ++g) {
        const int m = m_base + im * 16 + g + mrow_off;
        const int n = n_base + jn * 16 + ncol;
        const float v = acc[im][jn][g];
        if (mode == 2) {
          outf[(size_t)m * PROJ_DIM + n] = v;
        } else {
          const int b = m >> 11, s = m & 2047;   // m = b*S + s
          const int h = n >> 6,  d = n & 63;     // n = h*D + d
          size_t idx;
          if (mode == 0)
            idx = ((size_t)(b * HEAD_NUM + h) * CONTEXT_LEN + s) * HEAD_DIM + d;
          else
            idx = ((size_t)(b * HEAD_NUM + h) * HEAD_DIM + d) * CONTEXT_LEN + s;
          outh[idx] = (_Float16)v;
        }
      }
}

// ---------------------------------------------------------------------------
// Stage 2: causal flash attention. Block = 4 waves; wave owns 16 query rows.
// Q,K: [B][H][S][D] f16;  Vt: [B][H][D][S] f16;  ctx: [B][S][H*D] f16.
// ---------------------------------------------------------------------------
__global__ void __launch_bounds__(128)
flash_attn(const _Float16* __restrict__ Q, const _Float16* __restrict__ Kh,
           const _Float16* __restrict__ Vt, _Float16* __restrict__ ctx) {
  __shared__ _Float16 ldsP[4][16][64];   // per-wave P tile (C-layout -> A-frag)

  const int lane = threadIdx.x & 31;
  const int wid  = threadIdx.x >> 5;
  const int h = blockIdx.y, b = blockIdx.z;
  const int q0 = blockIdx.x * 64 + wid * 16;
  const int arow = lane & 15;
  const int mrow_off = (lane & 16) ? 8 : 0;

  const _Float16* Qb = Q  + (size_t)(b * HEAD_NUM + h) * CONTEXT_LEN * HEAD_DIM;
  const _Float16* Kb = Kh + (size_t)(b * HEAD_NUM + h) * CONTEXT_LEN * HEAD_DIM;
  const _Float16* Vb = Vt + (size_t)(b * HEAD_NUM + h) * HEAD_DIM * CONTEXT_LEN;

  v16h qf[2];
  qf[0] = load_a_frag(Qb + (size_t)(q0 + arow) * HEAD_DIM + 0,  lane);
  qf[1] = load_a_frag(Qb + (size_t)(q0 + arow) * HEAD_DIM + 32, lane);

  v8f zero = {};
  float mstate[8], lstate[8];
  v8f oacc[4];
#pragma unroll
  for (int g = 0; g < 8; ++g) { mstate[g] = -3.0e38f; lstate[g] = 0.0f; }
#pragma unroll
  for (int t = 0; t < 4; ++t) oacc[t] = zero;

  // causal: only key blocks with kb <= q0 contribute (kb mult of 64, q0 of 16)
  for (int kb = 0; kb <= q0; kb += 64) {
    // ---- scores 16x64 via 8 WMMAs
    v8f sacc[4];
#pragma unroll
    for (int nt = 0; nt < 4; ++nt) {
      sacc[nt] = zero;
#pragma unroll
      for (int ks = 0; ks < 2; ++ks) {
        v16h kf = load_b_frag(
            Kb + (size_t)(kb + nt * 16 + arow) * HEAD_DIM + ks * 32, lane);
        sacc[nt] = wmma_f16(qf[ks], kf, sacc[nt]);
      }
    }

    // ---- scale 1/sqrt(64), causal mask, per-row max
    const bool diag = (kb + 63 > q0);
    float sval[4][8], lm[8];
#pragma unroll
    for (int g = 0; g < 8; ++g) lm[g] = -3.0e38f;
#pragma unroll
    for (int nt = 0; nt < 4; ++nt)
#pragma unroll
      for (int g = 0; g < 8; ++g) {
        float v = sacc[nt][g] * 0.125f;
        if (diag) {
          const int key  = kb + nt * 16 + arow;
          const int qrow = q0 + g + mrow_off;
          if (key > qrow) v = -3.0e38f;
        }
        sval[nt][g] = v;
        lm[g] = fmaxf(lm[g], v);
      }
#pragma unroll
    for (int off = 1; off < 16; off <<= 1)
#pragma unroll
      for (int g = 0; g < 8; ++g)
        lm[g] = fmaxf(lm[g], __shfl_xor(lm[g], off, 32));

    // ---- online softmax update
    float rs[8];
#pragma unroll
    for (int g = 0; g < 8; ++g) {
      const float mn = fmaxf(mstate[g], lm[g]);
      const float sc = __expf(mstate[g] - mn);
      mstate[g] = mn;
      lstate[g] *= sc;
#pragma unroll
      for (int t = 0; t < 4; ++t) oacc[t][g] *= sc;
      rs[g] = 0.0f;
    }
#pragma unroll
    for (int nt = 0; nt < 4; ++nt)
#pragma unroll
      for (int g = 0; g < 8; ++g) {
        const float pv = __expf(sval[nt][g] - mstate[g]);
        rs[g] += pv;
        ldsP[wid][g + mrow_off][nt * 16 + arow] = (_Float16)pv;
      }
#pragma unroll
    for (int off = 1; off < 16; off <<= 1)
#pragma unroll
      for (int g = 0; g < 8; ++g) rs[g] += __shfl_xor(rs[g], off, 32);
#pragma unroll
    for (int g = 0; g < 8; ++g) lstate[g] += rs[g];

    // lane-crossing LDS round trip within one wave: drain DS before reload
    asm volatile("s_wait_dscnt 0" ::: "memory");

    // ---- P (A frags) x V (B frags): 8 WMMAs
    v16h pf[2];
    pf[0] = load_a_frag(&ldsP[wid][arow][0],  lane);
    pf[1] = load_a_frag(&ldsP[wid][arow][32], lane);
#pragma unroll
    for (int dt = 0; dt < 4; ++dt)
#pragma unroll
      for (int ks = 0; ks < 2; ++ks) {
        v16h vf = load_b_frag(
            Vb + (size_t)(dt * 16 + arow) * CONTEXT_LEN + kb + ks * 32, lane);
        oacc[dt] = wmma_f16(pf[ks], vf, oacc[dt]);
      }
  }

  // ---- normalize and store ctx [B][S][H*D]
#pragma unroll
  for (int dt = 0; dt < 4; ++dt)
#pragma unroll
    for (int g = 0; g < 8; ++g) {
      const int qrow = q0 + g + mrow_off;
      const int d = dt * 16 + arow;
      ctx[((size_t)b * CONTEXT_LEN + qrow) * PROJ_DIM + h * HEAD_DIM + d] =
          (_Float16)(oacc[dt][g] / lstate[g]);
    }
}

// ---------------------------------------------------------------------------
extern "C" void kernel_launch(void* const* d_in, const int* in_sizes, int n_in,
                              void* d_out, int out_size, void* d_ws,
                              size_t ws_size, hipStream_t stream) {
  (void)in_sizes; (void)n_in; (void)out_size; (void)ws_size;
  const float* X  = (const float*)d_in[0];
  const float* Wq = (const float*)d_in[1];
  const float* Wk = (const float*)d_in[2];
  const float* Wv = (const float*)d_in[3];
  const float* Wo = (const float*)d_in[4];
  float* out = (float*)d_out;

  // workspace layout (48 MB total)
  char* ws = (char*)d_ws;
  size_t off = 0;
  _Float16* Xh  = (_Float16*)(ws + off); off += (size_t)M_TOTAL * EMB_DIM * 2;
  _Float16* WqT = (_Float16*)(ws + off); off += (size_t)EMB_DIM * PROJ_DIM * 2;
  _Float16* WkT = (_Float16*)(ws + off); off += (size_t)EMB_DIM * PROJ_DIM * 2;
  _Float16* WvT = (_Float16*)(ws + off); off += (size_t)EMB_DIM * PROJ_DIM * 2;
  _Float16* WoT = (_Float16*)(ws + off); off += (size_t)EMB_DIM * PROJ_DIM * 2;
  _Float16* Qh  = (_Float16*)(ws + off); off += (size_t)M_TOTAL * PROJ_DIM * 2;
  _Float16* Kh  = (_Float16*)(ws + off); off += (size_t)M_TOTAL * PROJ_DIM * 2;
  _Float16* Vt  = (_Float16*)(ws + off); off += (size_t)M_TOTAL * PROJ_DIM * 2;
  _Float16* Ctx = (_Float16*)(ws + off); off += (size_t)M_TOTAL * PROJ_DIM * 2;

  const int nX = M_TOTAL * EMB_DIM;
  cvt_f32_to_f16<<<nX / 256, 256, 0, stream>>>(X, Xh, nX);
  const int nW = EMB_DIM * PROJ_DIM;
  transpose_cvt<<<nW / 256, 256, 0, stream>>>(Wq, WqT);
  transpose_cvt<<<nW / 256, 256, 0, stream>>>(Wk, WkT);
  transpose_cvt<<<nW / 256, 256, 0, stream>>>(Wv, WvT);
  transpose_cvt<<<nW / 256, 256, 0, stream>>>(Wo, WoT);

  dim3 gg(M_TOTAL / 256, PROJ_DIM / 128);
  gemm_wmma<<<gg, 256, 0, stream>>>(Xh, WqT, Qh, nullptr, 0);
  gemm_wmma<<<gg, 256, 0, stream>>>(Xh, WkT, Kh, nullptr, 0);
  gemm_wmma<<<gg, 256, 0, stream>>>(Xh, WvT, Vt, nullptr, 1);

  dim3 ga(CONTEXT_LEN / 64, HEAD_NUM, BATCH);
  flash_attn<<<ga, 128, 0, stream>>>(Qh, Kh, Vt, Ctx);

  gemm_wmma<<<gg, 256, 0, stream>>>(Ctx, WoT, nullptr, out, 2);
}
